// SwAVLoss_40355512713987
// MI455X (gfx1250) — matrix operand order repaired and checked by reading, not compiled, not run on hardware
//
#include <hip/hip_runtime.h>
#include <hip/hip_bf16.h>

// ---------------------------------------------------------------------------
// SwAV loss for MI455X (gfx1250, wave32).
//   Stage 0: convert z1/z2/W (f32) -> bf16 once (W zero-padded to 3072 rows).
//   Stage 1: scores = z @ W^T with bf16 WMMA (f32 accum), double-buffered
//            LDS tiles filled by GLOBAL_LOAD_ASYNC_TO_LDS_B128 (ASYNCcnt).
//   Stage 2: sinkhorn as diagonal scalings Q_ij = exp(s_ij/eps)*u_i*v_j.
//   Stage 3: loss via row-wise LSE + weighted reductions.
// ---------------------------------------------------------------------------

typedef __attribute__((ext_vector_type(16))) __bf16 bf16x16;
typedef __attribute__((ext_vector_type(8)))  __bf16 bf16x8;
typedef __attribute__((ext_vector_type(8)))  float  f32x8;

constexpr int Bn = 4096;   // batch
constexpr int Dn = 1024;   // feature dim
constexpr int Kn = 3000;   // prototypes
constexpr int Kp = 3008;   // padded row stride for scores (16B aligned rows)
constexpr int KnPad = 3072; // W rows padded to N-tile multiple (24*128)

constexpr float EPS_INV  = 20.0f;  // 1/0.05
constexpr float TEMP_INV = 10.0f;  // 1/0.1

// ------------------------ CDNA5 async-copy helpers -------------------------
__device__ __forceinline__ void async_ld_b128(unsigned lds_off, const void* g) {
  unsigned long long ga = (unsigned long long)g;
  asm volatile("global_load_async_to_lds_b128 %0, %1, off"
               :: "v"(lds_off), "v"(ga) : "memory");
}
__device__ __forceinline__ void wait_async0() {
  asm volatile("s_wait_asynccnt 0x0" ::: "memory");
}

// ------------------------------- GEMM --------------------------------------
// C[m,n] = sum_d Zc[m,d] * Wc[n,d]  (bf16 inputs), C stored f32 stride Kp.
// Block tile 128x128, 256 threads = 8 waves (2 x 4), wave tile 64x32.
constexpr int BM  = 128;
constexpr int BN_ = 128;
constexpr int BK  = 32;
constexpr int LDT = 40;    // LDS row stride in bf16 elements (32 + 8 pad)

__global__ __launch_bounds__(256)
void wmma_gemm_async(const __bf16* __restrict__ Zc, const __bf16* __restrict__ Wc,
                     float* __restrict__ out) {
  __shared__ __align__(16) __bf16 As[2][BM * LDT];   // 2 x 10 KB
  __shared__ __align__(16) __bf16 Bs[2][BN_ * LDT];  // 2 x 10 KB

  const int bn   = blockIdx.x;           // N tile (prototype rows)
  const int bm   = blockIdx.y;           // M tile (batch rows)
  const int tid  = threadIdx.x;
  const int wid  = tid >> 5;             // 8 waves
  const int lane = tid & 31;
  const int wm   = wid & 1;              // 0..1  -> 64-row slab
  const int wn   = wid >> 1;             // 0..3  -> 32-col slab
  const int lrow = lane & 15;
  const int khalfA = (lane >> 4) << 3;   // A operand: 0 or 8  (ISA A layout)
  const int kbaseB = (lane >> 4) << 3;   // B operand: 0 or 8  (ISA B layout)

  // staging: 128 rows x 4 chunks(16B) = 512 chunks per operand; 2 per thread
  const int srow0 = tid >> 2;            // chunk idx = tid      -> row, c
  const int sc0   = (tid & 3) << 3;      //  (c in bf16 elements: 0,8,16,24)
  const int srow1 = (tid + 256) >> 2;    // chunk idx = tid + 256
  const int sc1   = sc0;                 // (tid+256)&3 == tid&3

  const __bf16* Abase = Zc + (size_t)(bm * BM) * Dn;
  const __bf16* Bbase = Wc + (size_t)(bn * BN_) * Dn;

  f32x8 acc[4][2] = {};                  // 4 (M) x 2 (N) 16x16 tiles

  auto stage = [&](int buf, int k0) {
    async_ld_b128((unsigned)(size_t)&As[buf][srow0 * LDT + sc0],
                  Abase + (size_t)srow0 * Dn + k0 + sc0);
    async_ld_b128((unsigned)(size_t)&As[buf][srow1 * LDT + sc1],
                  Abase + (size_t)srow1 * Dn + k0 + sc1);
    async_ld_b128((unsigned)(size_t)&Bs[buf][srow0 * LDT + sc0],
                  Bbase + (size_t)srow0 * Dn + k0 + sc0);
    async_ld_b128((unsigned)(size_t)&Bs[buf][srow1 * LDT + sc1],
                  Bbase + (size_t)srow1 * Dn + k0 + sc1);
  };

  stage(0, 0);
  wait_async0();
  __syncthreads();

  int p = 0;
  for (int k0 = 0; k0 < Dn; k0 += BK) {
    if (k0 + BK < Dn) stage(p ^ 1, k0 + BK);   // prefetch next tile (async)

    bf16x16 afrag[4], bfrag[2];
#pragma unroll
    for (int mt = 0; mt < 4; ++mt) {
      int row = wm * 64 + mt * 16 + lrow;
      const bf16x8* q = reinterpret_cast<const bf16x8*>(&As[p][row * LDT + khalfA]);
      union { bf16x16 v; bf16x8 h[2]; } u;
      u.h[0] = q[0];        // K = khalf .. khalf+7
      u.h[1] = q[2];        // K = 16+khalf .. 16+khalf+7
      afrag[mt] = u.v;
    }
#pragma unroll
    for (int nt = 0; nt < 2; ++nt) {
      int row = wn * 32 + nt * 16 + lrow;
      const bf16x8* q = reinterpret_cast<const bf16x8*>(&Bs[p][row * LDT + kbaseB]);
      union { bf16x16 v; bf16x8 h[2]; } u;
      u.h[0] = q[0];        // K = kbase .. kbase+7
      u.h[1] = q[2];        // K = 16+kbase .. 16+kbase+7
      bfrag[nt] = u.v;
    }
#pragma unroll
    for (int mt = 0; mt < 4; ++mt)
#pragma unroll
      for (int nt = 0; nt < 2; ++nt)
        acc[mt][nt] = __builtin_amdgcn_wmma_f32_16x16x32_bf16(
            false, afrag[mt], false, bfrag[nt], (short)0, acc[mt][nt],
            false, false);

    wait_async0();          // my prefetch landed
    __syncthreads();        // everyone done reading buf p / writing buf p^1
    p ^= 1;
  }

  // ---- store (C/D layout: VGPR g -> M = g + 8*(lane>=16), N = lane%16) ----
  const int mhalf = (lane >> 4) << 3;
#pragma unroll
  for (int mt = 0; mt < 4; ++mt)
#pragma unroll
    for (int nt = 0; nt < 2; ++nt) {
      int col = bn * BN_ + wn * 32 + nt * 16 + lrow;
      if (col < Kn) {
        int rbase = bm * BM + wm * 64 + mt * 16 + mhalf;
#pragma unroll
        for (int g = 0; g < 8; ++g)
          out[(size_t)(rbase + g) * Kp + col] = acc[mt][nt][g];
      }
    }
}

// ------------------------- f32 -> bf16 conversion --------------------------
__global__ __launch_bounds__(256)
void cvt_f32_bf16(const float* __restrict__ in, __bf16* __restrict__ outp, int n4) {
  int i = blockIdx.x * 256 + threadIdx.x;
  if (i < n4) {
    const float4 f = reinterpret_cast<const float4*>(in)[i];
    __bf16* d = outp + (size_t)i * 4;
    d[0] = (__bf16)f.x; d[1] = (__bf16)f.y; d[2] = (__bf16)f.z; d[3] = (__bf16)f.w;
  }
}

__global__ __launch_bounds__(256)
void zero_bf16(__bf16* __restrict__ p, int n) {
  int i = blockIdx.x * 256 + threadIdx.x;
  if (i < n) p[i] = (__bf16)0.0f;
}

// --------------------------- block reductions ------------------------------
__device__ __forceinline__ float blockSum256(float v) {
  __shared__ float sm[256];
  int t = threadIdx.x;
  sm[t] = v; __syncthreads();
  for (int o = 128; o > 0; o >>= 1) {
    if (t < o) sm[t] += sm[t + o];
    __syncthreads();
  }
  float r = sm[0]; __syncthreads();
  return r;
}

__device__ __forceinline__ float blockMax256(float v) {
  __shared__ float sm[256];
  int t = threadIdx.x;
  sm[t] = v; __syncthreads();
  for (int o = 128; o > 0; o >>= 1) {
    if (t < o) sm[t] = fmaxf(sm[t], sm[t + o]);
    __syncthreads();
  }
  float r = sm[0]; __syncthreads();
  return r;
}

// ------------------------- sinkhorn helper kernels -------------------------
__global__ __launch_bounds__(256)
void row_expsum(const float* __restrict__ s, float* __restrict__ rp) {
  int i = blockIdx.x;
  const float* r = s + (size_t)i * Kp;
  float acc = 0.f;
  for (int j = threadIdx.x; j < Kn; j += 256) acc += __expf(r[j] * EPS_INV);
  acc = blockSum256(acc);
  if (threadIdx.x == 0) rp[i] = acc;
}

__global__ __launch_bounds__(256)
void init_u(const float* __restrict__ rp, float* __restrict__ u) {
  float acc = 0.f;
  for (int i = threadIdx.x; i < Bn; i += 256) acc += rp[i];
  float S = blockSum256(acc);
  float inv = 1.0f / S;
  for (int i = threadIdx.x; i < Bn; i += 256) u[i] = inv;
}

__global__ __launch_bounds__(256)
void col_update_v(const float* __restrict__ s, const float* __restrict__ u,
                  float* __restrict__ v) {
  int j = blockIdx.x * 256 + threadIdx.x;
  if (j >= Kn) return;
  float acc = 0.f;
  for (int i = 0; i < Bn; ++i)
    acc += __expf(s[(size_t)i * Kp + j] * EPS_INV) * u[i];
  v[j] = 1.0f / ((float)Kn * acc);
}

__global__ __launch_bounds__(256)
void row_update_u(const float* __restrict__ s, const float* __restrict__ v,
                  float* __restrict__ u) {
  int i = blockIdx.x;
  const float* r = s + (size_t)i * Kp;
  float acc = 0.f;
  for (int j = threadIdx.x; j < Kn; j += 256)
    acc += __expf(r[j] * EPS_INV) * v[j];
  acc = blockSum256(acc);
  if (threadIdx.x == 0) u[i] = 1.0f / ((float)Bn * acc);
}

__global__ __launch_bounds__(256)
void row_lse(const float* __restrict__ s, float* __restrict__ lse) {
  int i = blockIdx.x;
  const float* r = s + (size_t)i * Kp;
  float m = -3.0e38f;
  for (int j = threadIdx.x; j < Kn; j += 256) m = fmaxf(m, r[j]);
  m = blockMax256(m);
  float acc = 0.f;
  for (int j = threadIdx.x; j < Kn; j += 256)
    acc += __expf((r[j] - m) * TEMP_INV);
  acc = blockSum256(acc);
  if (threadIdx.x == 0) lse[i] = m * TEMP_INV + __logf(acc);
}

__global__ __launch_bounds__(256)
void loss_rows(const float* __restrict__ s1, const float* __restrict__ s2,
               const float* __restrict__ u1, const float* __restrict__ v1,
               const float* __restrict__ u2, const float* __restrict__ v2,
               const float* __restrict__ lse1, const float* __restrict__ lse2,
               float* __restrict__ part) {
  int i = blockIdx.x;
  const float* r1 = s1 + (size_t)i * Kp;
  const float* r2 = s2 + (size_t)i * Kp;
  float a = 0.f, qs2 = 0.f, b = 0.f, qs1 = 0.f;
  for (int j = threadIdx.x; j < Kn; j += 256) {
    float x1 = r1[j], x2 = r2[j];
    float e2 = __expf(x2 * EPS_INV) * v2[j];
    float e1 = __expf(x1 * EPS_INV) * v1[j];
    a   += e2 * x1;
    qs2 += e2;
    b   += e1 * x2;
    qs1 += e1;
  }
  a   = blockSum256(a);
  qs2 = blockSum256(qs2);
  b   = blockSum256(b);
  qs1 = blockSum256(qs1);
  if (threadIdx.x == 0) {
    float t = u2[i] * (a * TEMP_INV - qs2 * lse1[i]) +
              u1[i] * (b * TEMP_INV - qs1 * lse2[i]);
    part[i] = -t / (2.0f * (float)Bn);
  }
}

__global__ __launch_bounds__(256)
void final_sum(const float* __restrict__ part, float* __restrict__ out) {
  float acc = 0.f;
  for (int i = threadIdx.x; i < Bn; i += 256) acc += part[i];
  acc = blockSum256(acc);
  if (threadIdx.x == 0) out[0] = acc;
}

// ------------------------------- launcher ----------------------------------
extern "C" void kernel_launch(void* const* d_in, const int* in_sizes, int n_in,
                              void* d_out, int out_size, void* d_ws, size_t ws_size,
                              hipStream_t stream) {
  const float* z1 = (const float*)d_in[0];
  const float* z2 = (const float*)d_in[1];
  const float* Wm = (const float*)d_in[2];

  float* ws = (float*)d_ws;
  float* s1   = ws;                                  // [Bn, Kp]
  float* s2   = s1 + (size_t)Bn * Kp;                // [Bn, Kp]
  float* u1   = s2 + (size_t)Bn * Kp;                // [Bn]
  float* u2   = u1 + Bn;                             // [Bn]
  float* v1   = u2 + Bn;                             // [Kp]
  float* v2   = v1 + Kp;                             // [Kp]
  float* lse1 = v2 + Kp;                             // [Bn]
  float* lse2 = lse1 + Bn;                           // [Bn]
  float* rp   = lse2 + Bn;                           // [Bn]
  __bf16* zc1 = (__bf16*)(rp + Bn);                  // [Bn*Dn]   (16B aligned)
  __bf16* zc2 = zc1 + (size_t)Bn * Dn;               // [Bn*Dn]
  __bf16* Wc  = zc2 + (size_t)Bn * Dn;               // [KnPad*Dn]

  // ---- stage 0: one-time f32 -> bf16 conversion (W zero-padded) ----
  int n4z = Bn * Dn / 4;
  int n4w = Kn * Dn / 4;
  cvt_f32_bf16<<<(n4z + 255) / 256, 256, 0, stream>>>(z1, zc1, n4z);
  cvt_f32_bf16<<<(n4z + 255) / 256, 256, 0, stream>>>(z2, zc2, n4z);
  cvt_f32_bf16<<<(n4w + 255) / 256, 256, 0, stream>>>(Wm, Wc, n4w);
  int npad = (KnPad - Kn) * Dn;
  zero_bf16<<<(npad + 255) / 256, 256, 0, stream>>>(Wc + (size_t)Kn * Dn, npad);

  // ---- stage 1: WMMA GEMMs with async double-buffered LDS staging ----
  dim3 gg(KnPad / BN_, Bn / BM);                     // (24, 32)
  wmma_gemm_async<<<gg, 256, 0, stream>>>(zc1, Wc, s1);
  wmma_gemm_async<<<gg, 256, 0, stream>>>(zc2, Wc, s2);

  // ---- stage 2: sinkhorn (u,v diagonal scalings) ----
  for (int m = 0; m < 2; ++m) {
    float* s = m ? s2 : s1;
    float* u = m ? u2 : u1;
    float* v = m ? v2 : v1;
    row_expsum<<<Bn, 256, 0, stream>>>(s, rp);
    init_u<<<1, 256, 0, stream>>>(rp, u);
    for (int it = 0; it < 3; ++it) {
      col_update_v<<<(Kn + 255) / 256, 256, 0, stream>>>(s, u, v);
      row_update_u<<<Bn, 256, 0, stream>>>(s, v, u);
    }
  }

  // ---- stage 3: loss ----
  row_lse<<<Bn, 256, 0, stream>>>(s1, lse1);
  row_lse<<<Bn, 256, 0, stream>>>(s2, lse2);
  loss_rows<<<Bn, 256, 0, stream>>>(s1, s2, u1, v1, u2, v2, lse1, lse2, rp);
  final_sum<<<1, 256, 0, stream>>>(rp, (float*)d_out);
}
